// yolo_81750407512126
// MI455X (gfx1250) — matrix operand order repaired and checked by reading, not compiled, not run on hardware
//
#include <hip/hip_runtime.h>
#include <hip/hip_bf16.h>

// ---------------------------------------------------------------------------
// YOLO head on gfx1250: NHWC bf16 activations, BN folded into bf16 weights,
// convs = implicit GEMM on v_wmma_f32_16x16x32_bf16, double-buffered LDS fed
// by TENSOR_LOAD_TO_LDS (A/weight tiles) + global_load_async_to_lds_b128
// (B/im2col tiles); decode -> bitonic top-k -> greedy class-aware NMS.
// ---------------------------------------------------------------------------

typedef __bf16 bf16;
typedef __attribute__((ext_vector_type(16))) __bf16 v16bf;
typedef __attribute__((ext_vector_type(8)))  __bf16 v8bf;
typedef __attribute__((ext_vector_type(8)))  float  v8f;
typedef int v4i_ __attribute__((vector_size(16)));   // async builtin pointee
typedef unsigned int u32x4 __attribute__((ext_vector_type(4)));
typedef int i32x4 __attribute__((ext_vector_type(4)));
typedef int i32x8 __attribute__((ext_vector_type(8)));

#define BATCH   32
#define N_ALL   5415      // 3*19*19 + 3*38*38
#define K_PRE   1000
#define NPAD    8192
#define TM      64
#define TN      128
#define TK      32
#define LDA     40        // padded K-stride (bf16 elems) in LDS A tile
#define LDB     40        // padded K-stride (bf16 elems) in LDS B tile

#if defined(__has_builtin)
#if __has_builtin(__builtin_amdgcn_global_load_async_to_lds_b128) && \
    __has_builtin(__builtin_amdgcn_s_wait_asynccnt)
#define HAVE_ASYNC 1
#endif
#if __has_builtin(__builtin_amdgcn_tensor_load_to_lds) && \
    __has_builtin(__builtin_amdgcn_s_wait_tensorcnt)
#define HAVE_TDM 1
#endif
#endif

__device__ __forceinline__ void copy16_g2l(bf16* dst, const bf16* src) {
#ifdef HAVE_ASYNC
  __builtin_amdgcn_global_load_async_to_lds_b128(
      (__attribute__((address_space(1))) v4i_*)src,
      (__attribute__((address_space(3))) v4i_*)dst, 0, 0);
#else
  *(v8bf*)dst = *(const v8bf*)src;
#endif
}
__device__ __forceinline__ void zero16_l(bf16* dst) {
  const v8bf z = {(bf16)0.f, (bf16)0.f, (bf16)0.f, (bf16)0.f,
                  (bf16)0.f, (bf16)0.f, (bf16)0.f, (bf16)0.f};
  *(v8bf*)dst = z;
}
__device__ __forceinline__ void async_fence() {
#ifdef HAVE_ASYNC
  __builtin_amdgcn_s_wait_asynccnt(0);
#endif
}

// ------------------- BN fold + (r,s,ci)-reorder + quantize ------------------
// wq[co][rs*Cin+ci] = w[co][ci][r][s] * g/sqrt(v+eps);  bias = b - m*scale

__global__ void fold_bn_k(const float* __restrict__ w, const float* __restrict__ g,
                          const float* __restrict__ bb, const float* __restrict__ m,
                          const float* __restrict__ v,
                          bf16* __restrict__ wq, float* __restrict__ bias,
                          int Cout, int Cin, int ks) {
  const int K = Cin * ks * ks;
  int tid = blockIdx.x * blockDim.x + threadIdx.x;
  if (tid < Cout * K) {
    const int co = tid / K;
    const int k  = tid - co * K;
    const int rs = k / Cin;
    const int ci = k - rs * Cin;
    const int r  = rs / ks;
    const int s  = rs - r * ks;
    const float scale = g ? g[co] * rsqrtf(v[co] + 1e-5f) : 1.0f;
    wq[tid] = (bf16)(w[((co * Cin + ci) * ks + r) * ks + s] * scale);
  }
  if (tid < Cout) {
    if (g) {
      const float scale = g[tid] * rsqrtf(v[tid] + 1e-5f);
      bias[tid] = bb[tid] - m[tid] * scale;
    } else {
      bias[tid] = bb[tid];
    }
  }
}

// NCHW fp32 -> NHWC bf16
__global__ void cvt_nhwc_k(const float* __restrict__ x, bf16* __restrict__ y,
                           int C, int HW) {
  int tid = blockIdx.x * blockDim.x + threadIdx.x;
  if (tid >= BATCH * C * HW) return;
  const int c    = tid % C;
  const int rest = tid / C;
  const int hw   = rest % HW;
  const int b    = rest / HW;
  y[tid] = (bf16)x[((size_t)b * C + c) * HW + hw];
}

// --------------------------- WMMA implicit GEMM ----------------------------
// NHWC: X[(b*HW+hw)*Cin + ci].  M = Cout, N = BATCH*HW, K = Cin*KS*KS with
// k = rs*Cin + ci, so every 8-element K-chunk is contiguous in memory.

template <int KS>
__global__ __launch_bounds__(256, 1)
void conv_wmma_k(const bf16* __restrict__ Wq, const float* __restrict__ bias,
                 const bf16* __restrict__ X, bf16* __restrict__ Ybf,
                 float* __restrict__ Yf32, const bf16* __restrict__ Res,
                 int Cout, int Cin, int H, int W, int relu, int resW) {
  const int HW   = H * W;
  const int Ntot = BATCH * HW;
  const int Ktot = Cin * KS * KS;          // always a multiple of TK here

  __shared__ __align__(16) bf16 As[2][TM][LDA];
  __shared__ __align__(16) bf16 Bs[2][TN][LDB];

  const int tid  = threadIdx.x;
  const int lane = tid & 31;
  const int wave = tid >> 5;
  const int wm   = wave & 1;     // 2 waves along M
  const int wn   = wave >> 1;    // 4 waves along N
  const int half = lane >> 4;
  const int l16  = lane & 15;
  const int bm   = blockIdx.y * TM;
  const int bn   = blockIdx.x * TN;

#ifdef HAVE_TDM
  // ---- A tile via Tensor Data Mover: 64x32 bf16 2D tile, LDS-padded rows ---
  // D# group1 (constant per layer): data_size=2B, pad 16B after each 64B row,
  // tensor_dim0 = Ktot (elems), tensor_dim1 = rows remaining (zero-fill edge),
  // tile 32 x 64, dim0 stride = Ktot elems.  See CDNA5 ISA.
  const unsigned rows_left = (unsigned)(Cout - bm);
  i32x8 tdm_g1;
  tdm_g1[0] = (int)((1u << 16) | (1u << 20) | (3u << 22) | (3u << 25));
  tdm_g1[1] = (int)(((unsigned)Ktot & 0xFFFFu) << 16);
  tdm_g1[2] = (int)((((unsigned)Ktot >> 16) & 0xFFFFu) | ((rows_left & 0xFFFFu) << 16));
  tdm_g1[3] = (int)(((rows_left >> 16) & 0xFFFFu) | ((unsigned)TK << 16));
  tdm_g1[4] = TM;                       // tile_dim1=64, tile_dim2=0
  tdm_g1[5] = (int)(unsigned)Ktot;      // tensor_dim0_stride[31:0] (elems)
  tdm_g1[6] = 0;
  tdm_g1[7] = 0;
  const unsigned long long wq_addr = (unsigned long long)Wq;
  const unsigned lds_a0 =
      (unsigned)(unsigned long long)(__attribute__((address_space(3))) bf16*)&As[0][0][0];
  const unsigned lds_a1 =
      (unsigned)(unsigned long long)(__attribute__((address_space(3))) bf16*)&As[1][0][0];
  auto tdm_load_a = [&](int st, int k0) {
    const unsigned long long ga =
        wq_addr + ((unsigned long long)(unsigned)bm * (unsigned)Ktot + (unsigned)k0) * 2ull;
    u32x4 g0;
    g0[0] = 1u;                                         // count=1
    g0[1] = st ? lds_a1 : lds_a0;                       // lds_addr
    g0[2] = (unsigned)(ga & 0xFFFFFFFFull);             // global_addr[31:0]
    g0[3] = (unsigned)((ga >> 32) & 0x01FFFFFFull) | (2u << 30);  // [56:32] | type=2
    const i32x4 gz4 = {0, 0, 0, 0};
#if __clang_major__ >= 23
    const i32x8 gz8 = {0, 0, 0, 0, 0, 0, 0, 0};
    __builtin_amdgcn_tensor_load_to_lds(g0, tdm_g1, gz4, gz4, gz8, 0);
#else
    __builtin_amdgcn_tensor_load_to_lds(g0, tdm_g1, gz4, gz4, 0);
#endif
  };
#else
  // ---- A staging: one 16B chunk / thread, address math hoisted ----
  const int  a_row = tid >> 2;            // 0..63
  const int  a_kk  = (tid & 3) * 8;       // 0,8,16,24
  const bool a_ok  = (bm + a_row) < Cout;
  const bf16* a_src = Wq + (size_t)(bm + a_row) * Ktot + a_kk;
#endif

  // ---- B staging: two 16B chunks / thread, pixel decomposition hoisted ----
  int  b_kk[2], b_bb[2], b_oh[2], b_ow[2];
  const bf16* b_base[2];
  bool b_ok[2];
#pragma unroll
  for (int c = 0; c < 2; ++c) {
    const int cc = tid + c * 256;         // chunk id 0..511
    const int n  = cc >> 2;               // 0..127
    b_kk[c] = (cc & 3) * 8;
    const int gn = bn + n;
    b_ok[c] = gn < Ntot;
    const int gcl = b_ok[c] ? gn : 0;
    const int bb  = gcl / HW;
    const int hw  = gcl - bb * HW;
    b_bb[c] = bb;
    b_oh[c] = hw / W;
    b_ow[c] = hw - b_oh[c] * W;
    b_base[c] = X + ((size_t)bb * HW + hw) * Cin;
  }

  auto stage = [&](int st, int k0) {
#ifdef HAVE_TDM
    if (wave == 0) tdm_load_a(st, k0);    // one DMA per workgroup
#else
    { // A tile 64x32
      bf16* dst = &As[st][a_row][a_kk];
      if (a_ok) copy16_g2l(dst, a_src + k0);
      else      zero16_l(dst);
#ifndef HAVE_ASYNC
      if (a_ok && k0 + TK < Ktot)
        __builtin_prefetch(a_src + k0 + TK, 0, 3);   // global_prefetch_b8
#endif
    }
#endif
#pragma unroll
    for (int c = 0; c < 2; ++c) {         // B tile 128x32
      const int cc = tid + c * 256;
      bf16* dst = &Bs[st][cc >> 2][b_kk[c]];
      if (!b_ok[c]) { zero16_l(dst); continue; }
      const int gk = k0 + b_kk[c];
      if (KS == 1) {
        copy16_g2l(dst, b_base[c] + gk);
      } else {                            // 3x3, pad 1
        const int rs = gk / Cin;
        const int ci = gk - rs * Cin;
        const int r  = rs / 3 - 1;
        const int s  = rs - (rs / 3) * 3 - 1;
        const int ih = b_oh[c] + r;
        const int iw = b_ow[c] + s;
        if (ih >= 0 && ih < H && iw >= 0 && iw < W)
          copy16_g2l(dst, X + ((size_t)b_bb[c] * HW + ih * W + iw) * Cin + ci);
        else
          zero16_l(dst);
      }
    }
  };

  auto fence = [&]() {
#ifdef HAVE_TDM
    if (wave == 0) __builtin_amdgcn_s_wait_tensorcnt(0);
#endif
    async_fence();
    __syncthreads();
  };

  v8f acc[2][2];
  const v8f vzero = {0.f, 0.f, 0.f, 0.f, 0.f, 0.f, 0.f, 0.f};
#pragma unroll
  for (int i = 0; i < 2; ++i)
#pragma unroll
    for (int j = 0; j < 2; ++j) acc[i][j] = vzero;

  stage(0, 0);
  fence();

  const int nk = Ktot / TK;
  for (int kt = 0; kt < nk; ++kt) {
    const int cur = kt & 1;
    if (kt + 1 < nk) stage(cur ^ 1, (kt + 1) * TK);  // overlap next stage

    // fragments per CDNA5 ISA 16-bit layouts
    v16bf afrag[2], bfrag[2];
#pragma unroll
    for (int tm = 0; tm < 2; ++tm) {
      const bf16* p0 = &As[cur][wm * 32 + tm * 16 + l16][half * 8];
      v8bf a0 = *(const v8bf*)p0;
      v8bf a1 = *(const v8bf*)(p0 + 16);
#pragma unroll
      for (int i = 0; i < 8; ++i) { afrag[tm][i] = a0[i]; afrag[tm][i + 8] = a1[i]; }
    }
#pragma unroll
    for (int tn = 0; tn < 2; ++tn) {
      const bf16* q0 = &Bs[cur][wn * 32 + tn * 16 + l16][half * 8];
      v8bf b0 = *(const v8bf*)q0;
      v8bf b1 = *(const v8bf*)(q0 + 16);
#pragma unroll
      for (int i = 0; i < 8; ++i) { bfrag[tn][i] = b0[i]; bfrag[tn][i + 8] = b1[i]; }
    }
#pragma unroll
    for (int tm = 0; tm < 2; ++tm)
#pragma unroll
      for (int tn = 0; tn < 2; ++tn)
        acc[tm][tn] = __builtin_amdgcn_wmma_f32_16x16x32_bf16(
            false, afrag[tm], false, bfrag[tn], (short)0, acc[tm][tn], false, false);

    fence();
  }

  // ---- epilogue: bias + ReLU (+ nearest-2x residual), NHWC stores ----
#pragma unroll
  for (int tm = 0; tm < 2; ++tm) {
    const int gm0 = bm + wm * 32 + tm * 16 + half * 8;  // 8 consecutive Cout
#pragma unroll
    for (int tn = 0; tn < 2; ++tn) {
      const int gn = bn + wn * 32 + tn * 16 + l16;
      if (gn >= Ntot) continue;
      const int bb = gn / HW;
      const int hw = gn - bb * HW;
      float yv[8];
#pragma unroll
      for (int r = 0; r < 8; ++r) {
        float y = acc[tm][tn][r] + ((gm0 + r < Cout) ? bias[gm0 + r] : 0.f);
        if (relu) y = fmaxf(y, 0.f);
        yv[r] = y;
      }
      if (Res) {
        const int oh = hw / W, ow = hw - (hw / W) * W;
        const bf16* rp = Res +
            ((size_t)bb * (resW * resW) + (oh >> 1) * resW + (ow >> 1)) * Cout + gm0;
#pragma unroll
        for (int r = 0; r < 8; ++r)
          if (gm0 + r < Cout) yv[r] += (float)rp[r];
      }
      if (Ybf) {
        bf16* op = Ybf + ((size_t)bb * HW + hw) * Cout + gm0;
        if (gm0 + 8 <= Cout) {
          v8bf ov;
#pragma unroll
          for (int r = 0; r < 8; ++r) ov[r] = (bf16)yv[r];
          *(v8bf*)op = ov;                          // global_store_b128
        } else {
#pragma unroll
          for (int r = 0; r < 8; ++r)
            if (gm0 + r < Cout) op[r] = (bf16)yv[r];
        }
      }
      if (Yf32) {
        float* op = Yf32 + ((size_t)bb * HW + hw) * Cout + gm0;
#pragma unroll
        for (int r = 0; r < 8; ++r)
          if (gm0 + r < Cout) op[r] = yv[r];
      }
    }
  }
}

// ------------------------------ decode (NHWC) -------------------------------

__global__ void decode_k(const float* __restrict__ head,
                         float* __restrict__ boxes, float* __restrict__ objA,
                         float* __restrict__ clsS, float* __restrict__ clsI,
                         float* __restrict__ score,
                         int H, int W, float stride, int nOffset,
                         float aw0, float ah0, float aw1, float ah1, float aw2, float ah2) {
  const int HW = H * W;
  const int total = BATCH * 3 * HW;
  int tid = blockIdx.x * blockDim.x + threadIdx.x;
  if (tid >= total) return;
  const int b   = tid / (3 * HW);
  const int rem = tid - b * 3 * HW;
  const int a   = rem / HW;
  const int hw  = rem - a * HW;
  const int h   = hw / W;
  const int w   = hw - h * W;
  const float* p = head + ((size_t)b * HW + hw) * 75 + a * 25;  // 25 contiguous
  const float obj = 1.f / (1.f + __expf(-p[4]));
  float best = -1.f; int bestc = 0;
  for (int c = 0; c < 20; ++c) {
    float cl = 1.f / (1.f + __expf(-p[5 + c]));
    if (cl > best) { best = cl; bestc = c; }
  }
  const float aw = (a == 0) ? aw0 : ((a == 1) ? aw1 : aw2);
  const float ah = (a == 0) ? ah0 : ((a == 1) ? ah1 : ah2);
  const float cx = (1.f / (1.f + __expf(-p[0])) + (float)w) * stride;
  const float cy = (1.f / (1.f + __expf(-p[1])) + (float)h) * stride;
  const float bw = __expf(p[2]) * aw;
  const float bh = __expf(p[3]) * ah;
  const int row = b * N_ALL + nOffset + a * HW + hw;
  boxes[row * 4 + 0] = cx - bw * 0.5f;
  boxes[row * 4 + 1] = cy - bh * 0.5f;
  boxes[row * 4 + 2] = cx + bw * 0.5f;
  boxes[row * 4 + 3] = cy + bh * 0.5f;
  objA[row] = obj;
  clsS[row] = best;
  clsI[row] = (float)bestc;
  const float s = obj * best;
  score[row] = (s > 0.01f) ? s : 0.f;
}

// ------------------------- per-image bitonic top-k --------------------------

__global__ __launch_bounds__(256, 1)
void topk_k(const float* __restrict__ score, int* __restrict__ topIdx,
            float* __restrict__ topScore) {
  __shared__ unsigned long long key[NPAD];   // 64KB in 320KB LDS/WGP
  const int b = blockIdx.x, tid = threadIdx.x;
  for (int i = tid; i < NPAD; i += 256) {
    unsigned long long k = 0ull;
    if (i < N_ALL) {
      unsigned int hi = __float_as_uint(score[b * N_ALL + i]);  // score >= 0
      unsigned int lo = 0x7FFFFFFFu - (unsigned int)i;          // ties: low idx first
      k = ((unsigned long long)hi << 32) | lo;
    }
    key[i] = k;
  }
  __syncthreads();
  for (int kk = 2; kk <= NPAD; kk <<= 1) {
    for (int j = kk >> 1; j > 0; j >>= 1) {
      for (int i = tid; i < NPAD; i += 256) {
        const int ixj = i ^ j;
        if (ixj > i) {
          const bool desc = ((i & kk) == 0);
          unsigned long long a = key[i], c = key[ixj];
          if ((a < c) == desc) { key[i] = c; key[ixj] = a; }
        }
      }
      __syncthreads();
    }
  }
  for (int t = tid; t < K_PRE; t += 256) {
    const unsigned long long k = key[t];
    const unsigned int idx = 0x7FFFFFFFu - (unsigned int)k;
    topIdx[b * K_PRE + t]   = (idx < N_ALL) ? (int)idx : -1;
    topScore[b * K_PRE + t] = __uint_as_float((unsigned int)(k >> 32));
  }
}

// ------------------------ greedy class-aware NMS ----------------------------

__global__ __launch_bounds__(256, 1)
void nms_k(const float* __restrict__ boxes, const float* __restrict__ objA,
           const float* __restrict__ clsS, const float* __restrict__ clsI,
           const int* __restrict__ topIdx, const float* __restrict__ topScore,
           float* __restrict__ out) {
  __shared__ float bx1[K_PRE], by1[K_PRE], bx2[K_PRE], by2[K_PRE];
  __shared__ float sc[K_PRE], cid[K_PRE], ar[K_PRE], ob[K_PRE], cs[K_PRE];
  __shared__ unsigned char keep[K_PRE];
  const int b = blockIdx.x, tid = threadIdx.x;
  for (int t = tid; t < K_PRE; t += 256) {
    const int idx = topIdx[b * K_PRE + t];
    float x1 = 0, y1 = 0, x2 = 0, y2 = 0, o = 0, c = 0, ci = 0;
    if (idx >= 0) {
      const int r = (b * N_ALL + idx) * 4;
      x1 = boxes[r]; y1 = boxes[r + 1]; x2 = boxes[r + 2]; y2 = boxes[r + 3];
      o = objA[b * N_ALL + idx]; c = clsS[b * N_ALL + idx]; ci = clsI[b * N_ALL + idx];
    }
    bx1[t] = x1; by1[t] = y1; bx2[t] = x2; by2[t] = y2;
    ob[t] = o; cs[t] = c; cid[t] = ci;
    sc[t] = topScore[b * K_PRE + t];
    ar[t] = fmaxf(x2 - x1, 0.f) * fmaxf(y2 - y1, 0.f);
    keep[t] = 1;
  }
  for (int i = 0; i < K_PRE; ++i) {
    __syncthreads();
    if (!(keep[i] && sc[i] > 0.f)) continue;   // uniform (reads LDS)
    const float ax1 = bx1[i], ay1 = by1[i], ax2 = bx2[i], ay2 = by2[i];
    const float aa = ar[i], ac = cid[i];
    for (int j = i + 1 + tid; j < K_PRE; j += 256) {
      if (!keep[j] || cid[j] != ac) continue;
      const float ix1 = fmaxf(ax1, bx1[j]);
      const float iy1 = fmaxf(ay1, by1[j]);
      const float ix2 = fminf(ax2, bx2[j]);
      const float iy2 = fminf(ay2, by2[j]);
      const float inter = fmaxf(ix2 - ix1, 0.f) * fmaxf(iy2 - iy1, 0.f);
      const float iou = inter / (aa + ar[j] - inter + 1e-9f);
      if (iou > 0.45f) keep[j] = 0;
    }
  }
  __syncthreads();
  for (int t = tid; t < K_PRE; t += 256) {
    const float kf = (keep[t] && sc[t] > 0.f) ? 1.f : 0.f;
    float* o7 = out + ((size_t)b * K_PRE + t) * 7;
    o7[0] = bx1[t] * kf; o7[1] = by1[t] * kf; o7[2] = bx2[t] * kf; o7[3] = by2[t] * kf;
    o7[4] = ob[t] * kf;  o7[5] = cs[t] * kf;  o7[6] = cid[t] * kf;
  }
}

// --------------------------------- host ------------------------------------

static inline void* bump(char*& cur, size_t bytes) {
  void* p = (void*)cur;
  cur += (bytes + 255) & ~(size_t)255;
  return p;
}

extern "C" void kernel_launch(void* const* d_in, const int* in_sizes, int n_in,
                              void* d_out, int out_size, void* d_ws, size_t ws_size,
                              hipStream_t stream) {
  (void)in_sizes; (void)n_in; (void)out_size; (void)ws_size;
  const float* feat1 = (const float*)d_in[0];   // (32,96,38,38)   NCHW fp32
  const float* feat2 = (const float*)d_in[1];   // (32,1280,19,19) NCHW fp32
  const float* P[46];
  for (int i = 0; i < 46; ++i) P[i] = (const float*)d_in[i];

  char* cur = (char*)d_ws;
  const int HW19 = 19 * 19, HW38 = 38 * 38;
  // NHWC bf16 activation buffers
  bf16* b_feat2 = (bf16*)bump(cur, (size_t)BATCH * 1280 * HW19 * 2);
  bf16* b_feat1 = (bf16*)bump(cur, (size_t)BATCH * 96 * HW38 * 2);
  bf16* b_p5    = (bf16*)bump(cur, (size_t)BATCH * 512 * HW19 * 2);
  bf16* b_h51   = (bf16*)bump(cur, (size_t)BATCH * 512 * HW19 * 2);
  bf16* b_h52   = (bf16*)bump(cur, (size_t)BATCH * 1024 * HW19 * 2);
  bf16* b_up    = (bf16*)bump(cur, (size_t)BATCH * 256 * HW19 * 2);
  bf16* b_p41   = (bf16*)bump(cur, (size_t)BATCH * 96 * HW38 * 2);
  bf16* b_p4    = (bf16*)bump(cur, (size_t)BATCH * 256 * HW38 * 2);
  bf16* b_h41   = (bf16*)bump(cur, (size_t)BATCH * 256 * HW38 * 2);
  bf16* b_h42   = (bf16*)bump(cur, (size_t)BATCH * 512 * HW38 * 2);
  float* f_out0 = (float*)bump(cur, (size_t)BATCH * 75 * HW19 * 4);  // NHWC
  float* f_out1 = (float*)bump(cur, (size_t)BATCH * 75 * HW38 * 4);  // NHWC
  // folded weights (bf16, (rs,ci) K-order) + fp32 biases
  struct L { int co, ci, ks; };
  const L ls[10] = {{512,1280,1},{512,512,3},{1024,512,1},{75,1024,1},{256,512,1},
                    {96,96,3},{256,96,1},{256,256,3},{512,256,1},{75,512,1}};
  bf16*  wq[10]; float* bs[10];
  for (int i = 0; i < 10; ++i) {
    const size_t K = (size_t)ls[i].ci * ls[i].ks * ls[i].ks;
    wq[i] = (bf16*)bump(cur, (size_t)ls[i].co * K * 2);
    bs[i] = (float*)bump(cur, (size_t)ls[i].co * 4);
  }
  // decode buffers
  float* g_boxes = (float*)bump(cur, (size_t)BATCH * N_ALL * 4 * 4);
  float* g_obj   = (float*)bump(cur, (size_t)BATCH * N_ALL * 4);
  float* g_clsS  = (float*)bump(cur, (size_t)BATCH * N_ALL * 4);
  float* g_clsI  = (float*)bump(cur, (size_t)BATCH * N_ALL * 4);
  float* g_score = (float*)bump(cur, (size_t)BATCH * N_ALL * 4);
  int*   g_tIdx  = (int*)bump(cur, (size_t)BATCH * K_PRE * 4);
  float* g_tSc   = (float*)bump(cur, (size_t)BATCH * K_PRE * 4);

  // ---- fold BN + reorder + quantize weights ----
  const int pw[10] = {2, 7, 12, 17, 19, 24, 29, 34, 39, 44};
  const bool hasBN[10] = {true, true, true, false, true, true, true, true, true, false};
  for (int i = 0; i < 10; ++i) {
    const int K = ls[i].ci * ls[i].ks * ls[i].ks;
    const int total = ls[i].co * K;
    const int pb = pw[i];
    if (hasBN[i])
      fold_bn_k<<<(total + 255) / 256, 256, 0, stream>>>(
          P[pb], P[pb + 1], P[pb + 2], P[pb + 3], P[pb + 4],
          wq[i], bs[i], ls[i].co, ls[i].ci, ls[i].ks);
    else
      fold_bn_k<<<(total + 255) / 256, 256, 0, stream>>>(
          P[pb], nullptr, P[pb + 1], nullptr, nullptr,
          wq[i], bs[i], ls[i].co, ls[i].ci, ls[i].ks);
  }
  // ---- NCHW fp32 -> NHWC bf16 ----
  {
    int n2 = BATCH * 1280 * HW19;
    cvt_nhwc_k<<<(n2 + 255) / 256, 256, 0, stream>>>(feat2, b_feat2, 1280, HW19);
    int n1 = BATCH * 96 * HW38;
    cvt_nhwc_k<<<(n1 + 255) / 256, 256, 0, stream>>>(feat1, b_feat1, 96, HW38);
  }
  // ---- conv stack ----
  auto conv1 = [&](int li, const bf16* X, bf16* Ybf, float* Yf, const bf16* Res,
                   int H, int relu, int resW) {
    const int N = BATCH * H * H;
    dim3 grid((N + TN - 1) / TN, (ls[li].co + TM - 1) / TM);
    conv_wmma_k<1><<<grid, 256, 0, stream>>>(wq[li], bs[li], X, Ybf, Yf, Res,
                                             ls[li].co, ls[li].ci, H, H, relu, resW);
  };
  auto conv3 = [&](int li, const bf16* X, bf16* Ybf, int H) {
    const int N = BATCH * H * H;
    dim3 grid((N + TN - 1) / TN, (ls[li].co + TM - 1) / TM);
    conv_wmma_k<3><<<grid, 256, 0, stream>>>(wq[li], bs[li], X, Ybf, nullptr, nullptr,
                                             ls[li].co, ls[li].ci, H, H, 1, 0);
  };
  conv1(0, b_feat2, b_p5,  nullptr, nullptr, 19, 1, 0);   // conv_p5 1x1 1280->512
  conv3(1, b_p5,    b_h51, 19);                           // h5_1 3x3 512->512
  conv1(2, b_h51,   b_h52, nullptr, nullptr, 19, 1, 0);   // h5_2 1x1 512->1024
  conv1(3, b_h52,   nullptr, f_out0, nullptr, 19, 0, 0);  // h5_out 1x1 1024->75
  conv1(4, b_p5,    b_up,  nullptr, nullptr, 19, 1, 0);   // up 1x1 512->256
  conv3(5, b_feat1, b_p41, 38);                           // p4_1 3x3 96->96
  conv1(6, b_p41,   b_p4,  nullptr, b_up,    38, 1, 19);  // p4_2 1x1 96->256 (+2x up)
  conv3(7, b_p4,    b_h41, 38);                           // h4_1 3x3 256->256
  conv1(8, b_h41,   b_h42, nullptr, nullptr, 38, 1, 0);   // h4_2 1x1 256->512
  conv1(9, b_h42,   nullptr, f_out1, nullptr, 38, 0, 0);  // h4_out 1x1 512->75
  // ---- decode ----
  {
    int t0 = BATCH * 3 * HW19;
    decode_k<<<(t0 + 255) / 256, 256, 0, stream>>>(f_out0, g_boxes, g_obj, g_clsS, g_clsI,
        g_score, 19, 19, 32.f, 0, 81.f, 82.f, 135.f, 169.f, 344.f, 319.f);
    int t1 = BATCH * 3 * HW38;
    decode_k<<<(t1 + 255) / 256, 256, 0, stream>>>(f_out1, g_boxes, g_obj, g_clsS, g_clsI,
        g_score, 38, 38, 16.f, 3 * HW19, 10.f, 14.f, 23.f, 27.f, 37.f, 58.f);
  }
  // ---- top-k + NMS ----
  topk_k<<<BATCH, 256, 0, stream>>>(g_score, g_tIdx, g_tSc);
  nms_k<<<BATCH, 256, 0, stream>>>(g_boxes, g_obj, g_clsS, g_clsI, g_tIdx, g_tSc,
                                   (float*)d_out);
}
